// ContextLayer_56873956933987
// MI455X (gfx1250) — matrix operand order repaired and checked by reading, not compiled, not run on hardware
//
#include <hip/hip_runtime.h>
#include <hip/hip_bf16.h>

// ---- CDNA5 WMMA vector types ----
typedef __attribute__((ext_vector_type(16))) __bf16 v16bf;
typedef __attribute__((ext_vector_type(8)))  __bf16 v8bf;
typedef __attribute__((ext_vector_type(4)))  __bf16 v4bf;
typedef __attribute__((ext_vector_type(8)))  float  v8f;
typedef __attribute__((ext_vector_type(4)))  int    v4i;

#define NEGC (-100000.0f)

constexpr int B_ = 16, S_ = 4096, D_ = 768, Q_ = 64;

// ---- CDNA5 async global->LDS copy (ASYNCcnt path), guarded ----
#if defined(__HIP_DEVICE_COMPILE__) && __has_builtin(__builtin_amdgcn_global_load_async_to_lds_b128)
#define ASYNC_LDS 1
#endif

// pointer-to-v4i in explicit target address spaces (AS1 = global, AS3 = LDS)
typedef __attribute__((address_space(1))) v4i* as1_v4i_p;
typedef __attribute__((address_space(3))) v4i* as3_v4i_p;

__device__ __forceinline__ void copy16_g2l(const __bf16* g, __bf16* l) {
#ifdef ASYNC_LDS
    __builtin_amdgcn_global_load_async_to_lds_b128(
        (as1_v4i_p)(g), (as3_v4i_p)(l), 0, 0);
#else
    *(v8bf*)l = *(const v8bf*)g;
#endif
}
__device__ __forceinline__ void async_fence() {
#ifdef ASYNC_LDS
#if __has_builtin(__builtin_amdgcn_s_wait_asynccnt)
    __builtin_amdgcn_s_wait_asynccnt(0);
#else
    asm volatile("s_wait_asynccnt 0x0" ::: "memory");
#endif
#endif
}

__device__ __forceinline__ v4bf cvt4(float4 c) {
    v4bf r;
    r.x = (__bf16)c.x; r.y = (__bf16)c.y; r.z = (__bf16)c.z; r.w = (__bf16)c.w;
    return r;
}

// --------------------------------------------------------------------------
// Kernel 1: scores[b,q,s] = sum_d query[q,d] * ctx[b,s,d]  (+ mask), f32 out.
// Block = 256 thr (8 waves). Tile: 64 q x 128 s.
// Whole query (64x768 bf16, 96 KB) staged in LDS once; context double-buffered
// in 2 x 8 KB so each iteration needs one barrier and overlaps global loads
// with WMMA.
// --------------------------------------------------------------------------
__global__ __launch_bounds__(256)
void k_scores(const float* __restrict__ ctx, const float* __restrict__ mask,
              const float* __restrict__ query, float* __restrict__ scores)
{
    __shared__ __bf16 ldsQ[64 * D_];      // whole query, bf16, 96 KB
    __shared__ __bf16 ldsC[2][128 * 32];  // double-buffered ctx chunk

    const int t = threadIdx.x;
    const int lane = t & 31, wave = t >> 5;
    const int l16 = lane & 15, half = lane >> 4;
    const int sb = blockIdx.x * 128;
    const int b  = blockIdx.y;

    v8f acc[4] = {};

    // ---- one-time: whole query -> bf16 LDS ----
    for (int g = t; g < 64 * (D_ / 8); g += 256) {
        const int row = g / (D_ / 8);
        const int col = (g % (D_ / 8)) * 8;
        const float4* qp = (const float4*)(query + row * D_ + col);
        *(v4bf*)&ldsQ[row * D_ + col + 0] = cvt4(qp[0]);
        *(v4bf*)&ldsQ[row * D_ + col + 4] = cvt4(qp[1]);
    }

    // ---- ctx staging map: 128 rows x 2 groups of 16 cols ----
    const int cs = t >> 1, cg = t & 1;
    const float* cbase = ctx + ((size_t)b * S_ + sb + cs) * D_ + cg * 16;
    const int cl = cs * 32 + cg * 16;

    constexpr int NIT = D_ / 32;   // 24
    float4 creg[4];
#pragma unroll
    for (int j = 0; j < 4; ++j) creg[j] = ((const float4*)cbase)[j];
#pragma unroll
    for (int j = 0; j < 4; ++j) *(v4bf*)&ldsC[0][cl + j * 4] = cvt4(creg[j]);

    int p = 0;
    for (int i = 0; i < NIT; ++i) {
        __syncthreads();
        if (i + 1 < NIT) {               // issue next chunk's global loads early
            const float4* np = (const float4*)(cbase + (i + 1) * 32);
#pragma unroll
            for (int j = 0; j < 4; ++j) creg[j] = np[j];
        }
        // B tile: column n = s = wave*16+l16, K = 16*half + 0..15 contiguous
        v16bf bt = *(const v16bf*)&ldsC[p][(wave * 16 + l16) * 32 + half * 16];
#pragma unroll
        for (int qi = 0; qi < 4; ++qi) {
            union { v16bf v; v8bf h[2]; } au;
            au.h[0] = *(const v8bf*)&ldsQ[(qi * 16 + l16) * D_ + i * 32 + half * 16];
            au.h[1] = *(const v8bf*)&ldsQ[(qi * 16 + l16) * D_ + i * 32 + half * 16 + 8];
            acc[qi] = __builtin_amdgcn_wmma_f32_16x16x32_bf16(
                false, au.v, false, bt, (short)0, acc[qi], false, false);
        }
        if (i + 1 < NIT) {
#pragma unroll
            for (int j = 0; j < 4; ++j) *(v4bf*)&ldsC[p ^ 1][cl + j * 4] = cvt4(creg[j]);
        }
        p ^= 1;
    }

    const int sg = sb + wave * 16 + l16;
    const float add = (1.0f - mask[(size_t)b * S_ + sg]) * NEGC;
#pragma unroll
    for (int qi = 0; qi < 4; ++qi)
#pragma unroll
        for (int r = 0; r < 8; ++r) {
            int q = qi * 16 + r + 8 * half;   // C layout: M = r + 8*half
            scores[((size_t)b * Q_ + q) * S_ + sg] = acc[qi][r] + add;
        }
}

// --------------------------------------------------------------------------
// Kernel 2: row softmax over S=4096; one block (256 thr) per (b,q) row.
// --------------------------------------------------------------------------
__global__ __launch_bounds__(256)
void k_softmax(const float* __restrict__ scores, __bf16* __restrict__ w)
{
    __shared__ float red[8];
    const int t = threadIdx.x;
    const size_t row = blockIdx.x;
    const float4* src = (const float4*)(scores + row * S_);

    float4 v[4];
    float lmax = -3.4e38f;
#pragma unroll
    for (int i = 0; i < 4; ++i) {
        v[i] = src[t + i * 256];
        lmax = fmaxf(lmax, fmaxf(fmaxf(v[i].x, v[i].y), fmaxf(v[i].z, v[i].w)));
    }
#pragma unroll
    for (int o = 16; o > 0; o >>= 1) lmax = fmaxf(lmax, __shfl_xor(lmax, o, 32));
    if ((t & 31) == 0) red[t >> 5] = lmax;
    __syncthreads();
    float m = red[0];
#pragma unroll
    for (int i = 1; i < 8; ++i) m = fmaxf(m, red[i]);
    __syncthreads();

    float lsum = 0.0f;
#pragma unroll
    for (int i = 0; i < 4; ++i) {
        v[i].x = __expf(v[i].x - m); v[i].y = __expf(v[i].y - m);
        v[i].z = __expf(v[i].z - m); v[i].w = __expf(v[i].w - m);
        lsum += v[i].x + v[i].y + v[i].z + v[i].w;
    }
#pragma unroll
    for (int o = 16; o > 0; o >>= 1) lsum += __shfl_xor(lsum, o, 32);
    if ((t & 31) == 0) red[t >> 5] = lsum;
    __syncthreads();
    float s = 0.0f;
#pragma unroll
    for (int i = 0; i < 8; ++i) s += red[i];
    const float inv = __frcp_rn(s);

    __bf16* dst = w + row * S_;
#pragma unroll
    for (int i = 0; i < 4; ++i) {
        v4bf o;
        o.x = (__bf16)(v[i].x * inv); o.y = (__bf16)(v[i].y * inv);
        o.z = (__bf16)(v[i].z * inv); o.w = (__bf16)(v[i].w * inv);
        *(v4bf*)&dst[(t + i * 256) * 4] = o;
    }
}

// --------------------------------------------------------------------------
// Kernel 3: out[b,q,d] = sum_s w[b,q,s] * ctx[b,s,d].
// Block = 256 thr (8 waves). Tile: 64 q x 64 d. Double-buffered LDS; the bf16
// weight tiles use the CDNA5 async global->LDS path (no VGPR round trip);
// context is transposed+converted through VGPRs into LDS [d][s].
// --------------------------------------------------------------------------
__global__ __launch_bounds__(256)
void k_out(const float* __restrict__ ctx, const __bf16* __restrict__ w,
           float* __restrict__ out)
{
    __shared__ __bf16 ldsW[2][64 * 32];   // [q][s-chunk]
    __shared__ __bf16 ldsB[2][64 * 32];   // [d][s-chunk] (transposed ctx)

    const int t = threadIdx.x;
    const int lane = t & 31, wave = t >> 5;
    const int l16 = lane & 15, half = lane >> 4;
    const int db = blockIdx.x * 64;
    const int b  = blockIdx.y;
    const int di = wave & 3, qp = wave >> 2;

    v8f acc[2] = {};

    // staging maps
    const int wq = t >> 2, wg = t & 3;              // weights: 64 rows x 4 groups of 8
    const __bf16* wbase = w + ((size_t)b * Q_ + wq) * S_ + wg * 8;
    const int wl = wq * 32 + wg * 8;

    const int xs = t >> 3, xg = t & 7;              // ctx: 32 s-rows x 8 groups of 8 d
    const float* cbase = ctx + ((size_t)b * S_ + xs) * D_ + db + xg * 8;
    const int d0 = xg * 8;

    constexpr int NIT = S_ / 32;   // 128

    // ---- prologue: stage chunk 0 ----
    copy16_g2l(wbase, &ldsW[0][wl]);
    {
        const float4* cp = (const float4*)cbase;
        float4 c0 = cp[0], c1 = cp[1];
        ldsB[0][(d0 + 0) * 32 + xs] = (__bf16)c0.x;
        ldsB[0][(d0 + 1) * 32 + xs] = (__bf16)c0.y;
        ldsB[0][(d0 + 2) * 32 + xs] = (__bf16)c0.z;
        ldsB[0][(d0 + 3) * 32 + xs] = (__bf16)c0.w;
        ldsB[0][(d0 + 4) * 32 + xs] = (__bf16)c1.x;
        ldsB[0][(d0 + 5) * 32 + xs] = (__bf16)c1.y;
        ldsB[0][(d0 + 6) * 32 + xs] = (__bf16)c1.z;
        ldsB[0][(d0 + 7) * 32 + xs] = (__bf16)c1.w;
    }
    async_fence();

    int p = 0;
    for (int i = 0; i < NIT; ++i) {
        __syncthreads();
        float4 c0, c1;
        if (i + 1 < NIT) {
            // issue next chunk: async bf16 weights + ctx global loads
            copy16_g2l(wbase + (i + 1) * 32, &ldsW[p ^ 1][wl]);
            const float4* cp = (const float4*)(cbase + (size_t)(i + 1) * 32 * D_);
            c0 = cp[0]; c1 = cp[1];
        }
        // B tile: column n = d = di*16+l16, K = s = 16*half + 0..15 contiguous
        v16bf bt = *(const v16bf*)&ldsB[p][(di * 16 + l16) * 32 + half * 16];
#pragma unroll
        for (int j = 0; j < 2; ++j) {
            const int qi = qp * 2 + j;
            union { v16bf v; v8bf h[2]; } au;
            au.h[0] = *(const v8bf*)&ldsW[p][(qi * 16 + l16) * 32 + half * 16];
            au.h[1] = *(const v8bf*)&ldsW[p][(qi * 16 + l16) * 32 + half * 16 + 8];
            acc[j] = __builtin_amdgcn_wmma_f32_16x16x32_bf16(
                false, au.v, false, bt, (short)0, acc[j], false, false);
        }
        if (i + 1 < NIT) {
            __bf16* lb = &ldsB[p ^ 1][0];
            lb[(d0 + 0) * 32 + xs] = (__bf16)c0.x;
            lb[(d0 + 1) * 32 + xs] = (__bf16)c0.y;
            lb[(d0 + 2) * 32 + xs] = (__bf16)c0.z;
            lb[(d0 + 3) * 32 + xs] = (__bf16)c0.w;
            lb[(d0 + 4) * 32 + xs] = (__bf16)c1.x;
            lb[(d0 + 5) * 32 + xs] = (__bf16)c1.y;
            lb[(d0 + 6) * 32 + xs] = (__bf16)c1.z;
            lb[(d0 + 7) * 32 + xs] = (__bf16)c1.w;
            async_fence();   // weight copy must be LDS-visible before next barrier
        }
        p ^= 1;
    }

    const int dg = db + di * 16 + l16;
#pragma unroll
    for (int j = 0; j < 2; ++j) {
        const int qbase = (qp * 2 + j) * 16 + 8 * half;
#pragma unroll
        for (int r = 0; r < 8; ++r)
            out[((size_t)b * Q_ + (qbase + r)) * D_ + dg] = acc[j][r];
    }
}

// --------------------------------------------------------------------------
extern "C" void kernel_launch(void* const* d_in, const int* in_sizes, int n_in,
                              void* d_out, int out_size, void* d_ws, size_t ws_size,
                              hipStream_t stream)
{
    (void)in_sizes; (void)n_in; (void)out_size; (void)ws_size;
    const float* ctx   = (const float*)d_in[0];  // (B,S,D) f32
    const float* mask  = (const float*)d_in[1];  // (B,S)   f32
    const float* query = (const float*)d_in[2];  // (Q,D)   f32
    float* out = (float*)d_out;                  // (B,Q,D) f32

    float*  ws_scores = (float*)d_ws;                                   // B*Q*S f32
    __bf16* ws_w = (__bf16*)((char*)d_ws +
                             (size_t)B_ * Q_ * S_ * sizeof(float));     // B*Q*S bf16

    k_scores <<<dim3(S_ / 128, B_), 256, 0, stream>>>(ctx, mask, query, ws_scores);
    k_softmax<<<dim3(B_ * Q_),      256, 0, stream>>>(ws_scores, ws_w);
    k_out    <<<dim3(D_ / 64, B_),  256, 0, stream>>>(ctx, ws_w, out);
}
